// Discriminator_56135222558852
// MI455X (gfx1250) — matrix operand order repaired and checked by reading, not compiled, not run on hardware
//
#include <hip/hip_runtime.h>
#include <math.h>

// ---------------------------------------------------------------------------
// Discriminator forward pass for MI455X (gfx1250, wave32, WMMA).
// All GEMM operands live in f16 in global memory (weights converted+transposed
// once per launch; activation producers write f16 mirrors). GEMMs use
// v_wmma_f32_16x16x32_f16 with fp32 accumulate; LN/softmax/reductions fp32.
// Tile staging uses GLOBAL_LOAD_ASYNC_TO_LDS_B128 (ASYNCcnt path) when the
// toolchain exposes the builtin, else a register-staged sync copy.
// ---------------------------------------------------------------------------

#define BATCH 256
#define SEQ   150
#define SLD   152               // padded score row stride (16B-aligned in f16/f32)
#define DMODEL 512
#define NHEAD 8
#define DK    64
#define DFF   2048
#define TOK   (BATCH * SEQ)     // 38400

typedef __attribute__((ext_vector_type(16))) _Float16 v16h;
typedef __attribute__((ext_vector_type(8)))  _Float16 h8;
typedef __attribute__((ext_vector_type(8)))  float    v8f;

#if __has_builtin(__builtin_amdgcn_global_load_async_to_lds_b128) && \
    __has_builtin(__builtin_amdgcn_s_wait_asynccnt)
#define HAVE_ASYNC_LDS 1
typedef __attribute__((__vector_size__(4 * sizeof(int)))) int v4i_async;
#else
#define HAVE_ASYNC_LDS 0
#endif

// ---------------------------------------------------------------------------
// Generic strided-batched WMMA GEMM (all-f16 operands):
//   C = act( scale * (A @ B) + bias ),  written fp32 (Cf) and/or f16 (Ch)
//   A:  f16 row-major, (i,k) at A[aoff + i*lda + k]
//   B:  f16;  ncontig==1: (k,n) at B[boff + n*ldb + k]  (n-major, k contiguous)
//             ncontig==0: (k,n) at B[boff + k*ldb + n]  (k-major)
//   batch z -> (b = z/Hdiv, h = z%Hdiv); offsets b*s?b + h*s?h
// Tile: 128x64 per 256-thread block (8 wave32), K-step 32, 4 WMMA/wave/step,
//       double-buffered LDS, one barrier per step.
// ---------------------------------------------------------------------------
__global__ __launch_bounds__(256) void wmma_gemm_kernel(
    const _Float16* __restrict__ A, int lda, long sAb, long sAh,
    const _Float16* __restrict__ Bw, int ldb, long sBb, long sBh, int ncontig,
    const float* __restrict__ bias,
    float* __restrict__ Cf, _Float16* __restrict__ Ch,
    int ldc, long sCb, long sCh,
    int M, int N, int K, int Hdiv, float scale, int act)
{
    __shared__ _Float16 Ah[2][128][40];  // 40-half row stride: 80B, 16B aligned
    __shared__ _Float16 Bt[2][64][40];   // n-major (k contiguous per n)

    const int tid = threadIdx.x;
    const int z   = blockIdx.z;
    const int bz  = z / Hdiv;
    const int hz  = z % Hdiv;
    const long aoff = (long)bz * sAb + (long)hz * sAh;
    const long boff = (long)bz * sBb + (long)hz * sBh;
    const long coff = (long)bz * sCb + (long)hz * sCh;

    const int m0 = blockIdx.y * 128;
    const int n0 = blockIdx.x * 64;

    const int lane = tid & 31;
    const int wid  = tid >> 5;
    const int msub = wid & 3;         // M sub-tiles {msub, msub+4} (rows +0/+64)
    const int npr  = wid >> 2;        // 0/1 -> N sub-tiles {npr, npr+2}

    const int mrow0 = msub * 16 + (lane & 15);   // in 0..63
    const int kbA   = (lane >> 4) << 4;   // 0 or 16
    const int kbB   = (lane >> 4) << 4;   // 0 or 16

    // staging coordinates
    const int ar  = tid >> 1;         // A: row 0..127
    const int ac  = (tid & 1) << 4;   // A: k base {0,16}, 16 halves per thread
    const int bi  = tid >> 2;         // B ncontig: n 0..63
    const int bj  = (tid & 3) << 3;   // B ncontig: k base
    const int bkj = tid >> 3;         // B kmajor: k 0..31
    const int bn  = (tid & 7) << 3;   // B kmajor: n base

    const bool mfull = (m0 + 128 <= M);
    const bool nfull = (n0 + 64 <= N);
    const bool fA = mfull && ((lda & 7) == 0);
    const bool fB = nfull && ((ldb & 7) == 0);

    v8f acc00 = {0.f,0.f,0.f,0.f,0.f,0.f,0.f,0.f};
    v8f acc01 = acc00, acc10 = acc00, acc11 = acc00;

    h8 aR0 = {}, aR1 = {};
    h8 bR = {};
    int kpA = 0, kpB = 0;   // pending tile K offsets

    // ---- issue: start copy of tile kt into LDS[db] (async) or regs (sync) --
    auto stageA_issue = [&](int kt, int db) {
        kpA = kt;
        const int gm = m0 + ar;
        if (fA && (kt + 32 <= K)) {
            const _Float16* src = A + aoff + (long)gm * lda + kt + ac;
#if HAVE_ASYNC_LDS
            __builtin_amdgcn_global_load_async_to_lds_b128(
                (v4i_async*)src, (v4i_async*)&Ah[db][ar][ac], 0, 0);
            __builtin_amdgcn_global_load_async_to_lds_b128(
                (v4i_async*)(src + 8), (v4i_async*)&Ah[db][ar][ac + 8], 0, 0);
#else
            aR0 = *(const h8*)src;
            aR1 = *(const h8*)(src + 8);
#endif
        } else {
            const int  cm = (gm < M) ? gm : (M - 1);
            const long cb = aoff + (long)cm * lda;
            #pragma unroll
            for (int j = 0; j < 8; ++j) {
                const int gk0 = kt + ac + j;
                const int gk1 = gk0 + 8;
                const int ck0 = (gk0 < K) ? gk0 : (K - 1);
                const int ck1 = (gk1 < K) ? gk1 : (K - 1);
                const _Float16 v0 = A[cb + ck0];            // clamped, branchless
                const _Float16 v1 = A[cb + ck1];
                aR0[j] = (gm < M && gk0 < K) ? v0 : (_Float16)0.f;
                aR1[j] = (gm < M && gk1 < K) ? v1 : (_Float16)0.f;
            }
        }
        (void)db;
    };
    auto stageA_commit = [&](int db) {
#if HAVE_ASYNC_LDS
        if (fA && (kpA + 32 <= K)) return;                  // async wrote LDS
#endif
        *(h8*)(&Ah[db][ar][ac])     = aR0;
        *(h8*)(&Ah[db][ar][ac + 8]) = aR1;
    };
    auto stageB_issue = [&](int kt, int db) {
        kpB = kt;
        if (ncontig) {
            const int gn = n0 + bi;
            if (fB && (kt + 32 <= K)) {
                const _Float16* src = Bw + boff + (long)gn * ldb + kt + bj;
#if HAVE_ASYNC_LDS
                __builtin_amdgcn_global_load_async_to_lds_b128(
                    (v4i_async*)src, (v4i_async*)&Bt[db][bi][bj], 0, 0);
#else
                bR = *(const h8*)src;
#endif
            } else {
                const int  cn = (gn < N) ? gn : (N - 1);
                const long cb = boff + (long)cn * ldb;
                #pragma unroll
                for (int j = 0; j < 8; ++j) {
                    const int gk = kt + bj + j;
                    const int ck = (gk < K) ? gk : (K - 1);
                    const _Float16 v = Bw[cb + ck];
                    bR[j] = (gn < N && gk < K) ? v : (_Float16)0.f;
                }
            }
        } else {
            const int gk = kt + bkj;
            if (fB && (kt + 32 <= K)) {
                bR = *(const h8*)(Bw + boff + (long)gk * ldb + n0 + bn);
            } else {
                const int  ck = (gk < K) ? gk : (K - 1);
                const long cb = boff + (long)ck * ldb;
                #pragma unroll
                for (int j = 0; j < 8; ++j) {
                    const int gn = n0 + bn + j;
                    const int cn = (gn < N) ? gn : (N - 1);
                    const _Float16 v = Bw[cb + cn];
                    bR[j] = (gk < K && gn < N) ? v : (_Float16)0.f;
                }
            }
        }
        (void)db;
    };
    auto stageB_commit = [&](int db) {
        if (ncontig) {
#if HAVE_ASYNC_LDS
            if (fB && (kpB + 32 <= K)) return;              // async wrote LDS
#endif
            *(h8*)(&Bt[db][bi][bj]) = bR;
        } else {
            #pragma unroll
            for (int j = 0; j < 8; ++j)                     // transpose scatter
                Bt[db][bn + j][bkj] = bR[j];
        }
    };

    // -------- pipelined main loop: one barrier per K-step --------
    stageA_issue(0, 0);
    stageB_issue(0, 0);
    int db = 0;
    for (int k0 = 0; k0 < K; k0 += 32) {
        stageA_commit(db);
        stageB_commit(db);
#if HAVE_ASYNC_LDS
        __builtin_amdgcn_s_wait_asynccnt(0);
#endif
        __syncthreads();

        const int kn = k0 + 32;
        if (kn < K) { stageA_issue(kn, db ^ 1); stageB_issue(kn, db ^ 1); }

        v16h afrag0, afrag1;
        #pragma unroll
        for (int h = 0; h < 16; ++h) {
            const int kk = kbA + h;
            afrag0[h] = Ah[db][mrow0][kk];
            afrag1[h] = Ah[db][mrow0 + 64][kk];
        }

        v16h bfrag0, bfrag1;
        {
            const int nc0 = npr * 16 + (lane & 15);
            const int nc1 = nc0 + 32;
            #pragma unroll
            for (int h = 0; h < 16; ++h) {
                bfrag0[h] = Bt[db][nc0][kbB + h];
                bfrag1[h] = Bt[db][nc1][kbB + h];
            }
        }

        acc00 = __builtin_amdgcn_wmma_f32_16x16x32_f16(
                    false, afrag0, false, bfrag0, (short)0, acc00, false, false);
        acc01 = __builtin_amdgcn_wmma_f32_16x16x32_f16(
                    false, afrag0, false, bfrag1, (short)0, acc01, false, false);
        acc10 = __builtin_amdgcn_wmma_f32_16x16x32_f16(
                    false, afrag1, false, bfrag0, (short)0, acc10, false, false);
        acc11 = __builtin_amdgcn_wmma_f32_16x16x32_f16(
                    false, afrag1, false, bfrag1, (short)0, acc11, false, false);
        db ^= 1;
    }

    // -------- epilogue: scale, bias, leaky-relu, store fp32/f16 --------
    #pragma unroll
    for (int am = 0; am < 2; ++am) {
        #pragma unroll
        for (int an = 0; an < 2; ++an) {
            v8f acc = am ? (an ? acc11 : acc10) : (an ? acc01 : acc00);
            const int nsub = npr + (an << 1);
            const int mbase = m0 + msub * 16 + (am << 6);
            #pragma unroll
            for (int r = 0; r < 8; ++r) {
                const int gm = mbase + r + ((lane >> 4) << 3);
                const int gn = n0 + nsub * 16 + (lane & 15);
                if (gm < M && gn < N) {
                    float v = acc[r] * scale;
                    if (bias) v += bias[gn];
                    if (act)  v = v > 0.f ? v : 0.01f * v;
                    const long o = coff + (long)gm * ldc + gn;
                    if (Cf) Cf[o] = v;
                    if (Ch) Ch[o] = (_Float16)v;
                }
            }
        }
    }
}

// ---------------------------------------------------------------------------
// Weight convert + transpose to n-major f16:  wt[n*Kd + k] = (f16) w[k*Nd + n]
// ---------------------------------------------------------------------------
__global__ __launch_bounds__(256) void wconv_kernel(
    const float* __restrict__ w, _Float16* __restrict__ wt,
    int Kd, int Nd, long total)
{
    const long idx = (long)blockIdx.x * blockDim.x + threadIdx.x;
    if (idx >= total) return;
    const int  k = (int)(idx % Kd);
    const long n = idx / Kd;
    wt[idx] = (_Float16)w[(long)k * Nd + n];
}

// ---------------------------------------------------------------------------
// LayerNorm with residual: y = LN(a + x) * g + b  (D=512, 1 wave per row).
// Writes fp32 y and optional f16 mirror yh. y may alias x.
// ---------------------------------------------------------------------------
__global__ __launch_bounds__(256) void ln_kernel(
    float* __restrict__ y, _Float16* __restrict__ yh,
    const float* __restrict__ a, const float* __restrict__ x,
    const float* __restrict__ g, const float* __restrict__ b, int rows)
{
    const int row  = blockIdx.x * (blockDim.x >> 5) + (threadIdx.x >> 5);
    const int lane = threadIdx.x & 31;
    if (row >= rows) return;
    const float* pa = a + (long)row * DMODEL;
    const float* px = x + (long)row * DMODEL;
    float vals[16];
    float s = 0.f;
    #pragma unroll
    for (int i = 0; i < 16; ++i) {
        const int c = lane + (i << 5);
        const float v = pa[c] + px[c];
        vals[i] = v; s += v;
    }
    #pragma unroll
    for (int o = 16; o > 0; o >>= 1) s += __shfl_xor(s, o, 32);
    const float mean = s * (1.f / DMODEL);
    float s2 = 0.f;
    #pragma unroll
    for (int i = 0; i < 16; ++i) { const float d = vals[i] - mean; s2 += d * d; }
    #pragma unroll
    for (int o = 16; o > 0; o >>= 1) s2 += __shfl_xor(s2, o, 32);
    const float r = rsqrtf(s2 * (1.f / DMODEL) + 1e-5f);
    float*     py = y  + (long)row * DMODEL;
    _Float16*  ph = yh ? yh + (long)row * DMODEL : nullptr;
    #pragma unroll
    for (int i = 0; i < 16; ++i) {
        const int c = lane + (i << 5);
        const float o = (vals[i] - mean) * r * g[c] + b[c];
        py[c] = o;
        if (ph) ph[c] = (_Float16)o;
    }
}

// ---------------------------------------------------------------------------
// Softmax over rows of length 150: fp32 in (stride ldi) -> f16 out (stride ldo)
// ---------------------------------------------------------------------------
__global__ __launch_bounds__(256) void softmax150_kernel(
    const float* __restrict__ pin, _Float16* __restrict__ pout,
    int rows, int ldi, int ldo)
{
    const int row  = blockIdx.x * (blockDim.x >> 5) + (threadIdx.x >> 5);
    const int lane = threadIdx.x & 31;
    if (row >= rows) return;
    const float* pr = pin  + (long)row * ldi;
    _Float16*    po = pout + (long)row * ldo;
    float v[5];
    float mx = -1e30f;
    #pragma unroll
    for (int i = 0; i < 5; ++i) {
        const int c = lane + (i << 5);
        v[i] = (c < SEQ) ? pr[c] : -1e30f;
        mx = fmaxf(mx, v[i]);
    }
    #pragma unroll
    for (int o = 16; o > 0; o >>= 1) mx = fmaxf(mx, __shfl_xor(mx, o, 32));
    float s = 0.f;
    #pragma unroll
    for (int i = 0; i < 5; ++i) {
        const int c = lane + (i << 5);
        const float e = (c < SEQ) ? __expf(v[i] - mx) : 0.f;
        v[i] = e; s += e;
    }
    #pragma unroll
    for (int o = 16; o > 0; o >>= 1) s += __shfl_xor(s, o, 32);
    const float inv = 1.f / s;
    #pragma unroll
    for (int i = 0; i < 5; ++i) {
        const int c = lane + (i << 5);
        if (c < SEQ) po[c] = (_Float16)(v[i] * inv);
    }
}

// ---------------------------------------------------------------------------
// Per-batch normalization + shifted channel -> x5 (B,150,5). 1 wave per batch.
// ---------------------------------------------------------------------------
__global__ __launch_bounds__(32) void prep_kernel(
    const float* __restrict__ mv, const float* __restrict__ qr,
    float* __restrict__ x5)
{
    const int b    = blockIdx.x;
    const int lane = threadIdx.x;
    const float* mb = mv + (long)b * 480;   // (120,4)
    const float* qb = qr + (long)b * 120;   // (30,4)
    float s = 0.f, s2 = 0.f;
    for (int i = lane; i < 600; i += 32) {
        const float v = (i < 480) ? mb[i] : qb[i - 480];
        s += v; s2 += v * v;
    }
    #pragma unroll
    for (int o = 16; o > 0; o >>= 1) { s += __shfl_xor(s, o, 32); s2 += __shfl_xor(s2, o, 32); }
    const float mean = s * (1.f / 600.f);
    float var = (s2 - 600.f * mean * mean) * (1.f / 599.f);   // ddof=1
    var = fmaxf(var, 0.f);
    const float inv = 1.f / (sqrtf(var) + 1e-9f);
    float* xb = x5 + (long)b * (SEQ * 5);
    for (int i = lane; i < 600; i += 32) {
        const float v = (i < 480) ? mb[i] : qb[i - 480];
        const int t = i >> 2, c = i & 3;
        xb[t * 5 + 1 + c] = (v - mean) * inv;
    }
    for (int t = lane; t < SEQ; t += 32) {
        float raw;
        if (t == 0) raw = mb[0];                          // inputs[b,0,0]
        else {
            const int tt = t - 1;                         // inputs[b,t-1,3]
            raw = (tt < 120) ? mb[tt * 4 + 3] : qb[(tt - 120) * 4 + 3];
        }
        xb[t * 5] = (raw - mean) * inv;
    }
}

// ---------------------------------------------------------------------------
// Tiny K=5 input embedding: x = x5 @ W(5,512) + b   (dual fp32 + f16 write)
// ---------------------------------------------------------------------------
__global__ __launch_bounds__(256) void embed_kernel(
    const float* __restrict__ x5, const float* __restrict__ w,
    const float* __restrict__ bias, float* __restrict__ x,
    _Float16* __restrict__ xh, long total)
{
    const long idx = (long)blockIdx.x * blockDim.x + threadIdx.x;
    if (idx >= total) return;
    const int  n   = (int)(idx & (DMODEL - 1));
    const long row = idx >> 9;
    const float* r5 = x5 + row * 5;
    float acc = bias[n];
    #pragma unroll
    for (int c = 0; c < 5; ++c) acc += r5[c] * w[c * DMODEL + n];
    x[idx]  = acc;
    xh[idx] = (_Float16)acc;
}

__global__ __launch_bounds__(256) void addpos_kernel(
    float* __restrict__ x, _Float16* __restrict__ xh,
    const float* __restrict__ pos, long total)
{
    const long idx = (long)blockIdx.x * blockDim.x + threadIdx.x;
    if (idx >= total) return;
    const float v = x[idx] + pos[idx % ((long)SEQ * DMODEL)];
    x[idx]  = v;
    xh[idx] = (_Float16)v;
}

__global__ __launch_bounds__(256) void fill0_kernel(
    float* __restrict__ p, _Float16* __restrict__ ph, long n)
{
    const long idx = (long)blockIdx.x * blockDim.x + threadIdx.x;
    if (idx < n) { p[idx] = 0.f; ph[idx] = (_Float16)0.f; }
}

// ---------------------------------------------------------------------------
// Decoder cross-attention (query length 1): scores, context (small GEMVs).
// ---------------------------------------------------------------------------
__global__ __launch_bounds__(256) void dec_scores_kernel(
    const float* __restrict__ qc,       // (B,512) fp32
    const _Float16* __restrict__ kc,    // (B*150,512) f16
    float* __restrict__ sc)             // (B*H,150) fp32
{
    const int idx = blockIdx.x * blockDim.x + threadIdx.x;
    if (idx >= BATCH * NHEAD * SEQ) return;
    const int j  = idx % SEQ;
    const int bh = idx / SEQ;
    const int h  = bh % NHEAD;
    const int b  = bh / NHEAD;
    const float*    qp = qc + (long)b * DMODEL + h * DK;
    const _Float16* kp = kc + ((long)b * SEQ + j) * DMODEL + h * DK;
    float s = 0.f;
    #pragma unroll 8
    for (int d = 0; d < DK; ++d) s += qp[d] * (float)kp[d];
    sc[idx] = s * 0.125f;   // 1/sqrt(64)
}

__global__ __launch_bounds__(256) void dec_ctx_kernel(
    const _Float16* __restrict__ sc,    // (B*H,150) f16 probs
    const _Float16* __restrict__ vc,    // (B*150,512) f16
    _Float16* __restrict__ ctx)         // (B,512) f16
{
    const int idx = blockIdx.x * blockDim.x + threadIdx.x;
    if (idx >= BATCH * DMODEL) return;
    const int d = idx % DK;
    const int h = (idx / DK) % NHEAD;
    const int b = idx / DMODEL;
    const _Float16* pp = sc + ((long)b * NHEAD + h) * SEQ;
    const _Float16* vp = vc + (long)b * SEQ * DMODEL + h * DK + d;
    float s = 0.f;
    for (int j = 0; j < SEQ; ++j) s += (float)pp[j] * (float)vp[(long)j * DMODEL];
    ctx[(long)b * DMODEL + h * DK + d] = (_Float16)s;
}

// ---------------------------------------------------------------------------
// Final head: sigmoid(d @ w(512,1) + b)
// ---------------------------------------------------------------------------
__global__ __launch_bounds__(256) void head_kernel(
    const float* __restrict__ d, const float* __restrict__ w,
    const float* __restrict__ b, float* __restrict__ out)
{
    const int bi = blockIdx.x * blockDim.x + threadIdx.x;
    if (bi >= BATCH) return;
    const float* r = d + (long)bi * DMODEL;
    float s = b[0];
    #pragma unroll 8
    for (int c = 0; c < DMODEL; ++c) s += r[c] * w[c];
    out[bi] = 1.f / (1.f + expf(-s));
}

// ---------------------------------------------------------------------------
// Host side
// ---------------------------------------------------------------------------
struct Lin  { const float* w; const float* b; };
struct LNP  { const float* g; const float* b; };
struct Fcr  { Lin fi, ff; LNP ln; };
struct Attn { Lin WQ, WK, WV, WO; };

struct LinH  { const _Float16* wt; const float* b; };   // n-major f16 weight
struct FcrH  { LinH fi, ff; LNP ln; };
struct AttnH { LinH WQ, WK, WV, WO; };

static void gemm(hipStream_t st,
                 const _Float16* A, int lda, long sAb, long sAh,
                 const _Float16* Bw, int ldb, long sBb, long sBh, int ncontig,
                 const float* bias,
                 float* Cf, _Float16* Ch, int ldc, long sCb, long sCh,
                 int M, int N, int K, int batch, int Hdiv,
                 float scale, int act)
{
    dim3 g((N + 63) / 64, (M + 127) / 128, batch);
    wmma_gemm_kernel<<<g, 256, 0, st>>>(A, lda, sAb, sAh, Bw, ldb, sBb, sBh,
                                        ncontig, bias, Cf, Ch, ldc, sCb, sCh,
                                        M, N, K, Hdiv, scale, act);
}

static void run_ln(hipStream_t st, float* y, _Float16* yh,
                   const float* a, const float* x, const LNP& p, int rows)
{
    ln_kernel<<<(rows + 7) / 8, 256, 0, st>>>(y, yh, a, x, p.g, p.b, rows);
}

// fcres: h = leaky(xh@fi+b) [f16], a = leaky(h@ff+b) [fp32], x=LN(a+x) dual
static void run_fcres(hipStream_t st, float* x, _Float16* xh,
                      _Float16* tH, float* tA, const FcrH& p, int rows)
{
    gemm(st, xh, DMODEL, 0, 0, p.fi.wt, DMODEL, 0, 0, 1, p.fi.b,
         nullptr, tH, DFF, 0, 0, rows, DFF, DMODEL, 1, 1, 1.f, 1);
    gemm(st, tH, DFF, 0, 0, p.ff.wt, DFF, 0, 0, 1, p.ff.b,
         tA, nullptr, DMODEL, 0, 0, rows, DMODEL, DFF, 1, 1, 1.f, 1);
    run_ln(st, x, xh, tA, x, p.ln, rows);
}

extern "C" void kernel_launch(void* const* d_in, const int* in_sizes, int n_in,
                              void* d_out, int out_size, void* d_ws, size_t ws_size,
                              hipStream_t stream)
{
    (void)in_sizes; (void)n_in; (void)out_size; (void)ws_size;

    const float* mv = (const float*)d_in[0];   // (256,120,4)
    const float* qr = (const float*)d_in[1];   // (256,30,4)

    // ---- unpack params: JAX pytree leaf order (dict keys sorted) ----------
    int pi = 2;
    auto nxt = [&](void) -> const float* { return (const float*)d_in[pi++]; };
    auto rdLin  = [&](void) -> Lin  { Lin  l; l.b = nxt(); l.w = nxt(); return l; };
    auto rdLN   = [&](void) -> LNP  { LNP  l; l.b = nxt(); l.g = nxt(); return l; };
    auto rdFcr  = [&](void) -> Fcr  { Fcr  f; f.ff = rdLin(); f.fi = rdLin(); f.ln = rdLN(); return f; };
    auto rdAttn = [&](void) -> Attn { Attn a; a.WK = rdLin(); a.WO = rdLin();
                                      a.WQ = rdLin(); a.WV = rdLin(); return a; };

    // params sorted: bar, dec, enc, fc_disc, fcres, pos
    Lin bar_init; Fcr bar_res[2];
    { bar_init = rdLin(); bar_res[0] = rdFcr(); bar_res[1] = rdFcr(); }

    struct DecL { Attn cross; Fcr fcr; LNP ln1, ln2; Attn self; } dec[4];
    for (int i = 0; i < 4; ++i) {
        dec[i].cross = rdAttn(); dec[i].fcr = rdFcr();
        dec[i].ln1 = rdLN(); dec[i].ln2 = rdLN(); dec[i].self = rdAttn();
    }
    struct EncL { Attn attn; Fcr fcr; LNP ln; } enc[4];
    for (int i = 0; i < 4; ++i) {
        enc[i].attn = rdAttn(); enc[i].fcr = rdFcr(); enc[i].ln = rdLN();
    }
    Lin fc_disc = rdLin();
    Fcr fcres[4];
    for (int i = 0; i < 4; ++i) fcres[i] = rdFcr();
    const float* pos = nxt();

    // ---- workspace layout --------------------------------------------------
    const long TD  = (long)TOK * DMODEL;              // 19,660,800
    const long SCF = (long)BATCH * NHEAD * SEQ * SLD; // 46,694,400
    float* W = (float*)d_ws;
    float* xbuf = W;                          // TOK x 512 fp32
    float* abuf = xbuf + TD;                  // TOK x 512 fp32 (pre-LN outputs)
    float* scoresF = abuf + TD;               // attention scores fp32 (SLD stride)
    float* dout = scoresF + SCF;              // B x 512 fp32 decoder state
    float* dtmp = dout + (long)BATCH * DMODEL;
    float* dtm2 = dtmp + (long)BATCH * DMODEL;
    float* dsc  = dtm2 + (long)BATCH * DMODEL;        // B*H x 150 fp32
    float* x5   = dsc + (long)BATCH * NHEAD * SEQ;    // B x 150 x 5

    _Float16* xh  = (_Float16*)(x5 + (long)BATCH * SEQ * 5 + 3);  // pad
    xh = (_Float16*)((((size_t)xh) + 15) & ~(size_t)15);
    _Float16* qh  = xh + TD;                  // q, then attention ctx
    _Float16* kh  = qh + TD;
    _Float16* vh  = kh + TD;
    _Float16* hh  = vh + TD;                  // TOK x DFF f16 (aliases probs)
    _Float16* probsH = hh;                    // B*H*SEQ*SLD <= TOK*DFF
    _Float16* douth = hh + (long)TOK * DFF;   // B x 512
    _Float16* dtmph = douth + (long)BATCH * DMODEL;
    _Float16* dhh   = dtmph + (long)BATCH * DMODEL;   // B x 2048
    _Float16* dscH  = dhh + (long)BATCH * DFF;        // B*H x 150
    _Float16* wtcur = dscH + (long)BATCH * NHEAD * SEQ;

    // ---- convert all GEMM weights to n-major f16 ---------------------------
    auto conv = [&](const Lin& l, int Kd, int Nd) -> LinH {
        LinH o; o.b = l.b; o.wt = wtcur;
        const long total = (long)Kd * Nd;
        wconv_kernel<<<(int)((total + 255) / 256), 256, 0, stream>>>(
            l.w, wtcur, Kd, Nd, total);
        wtcur += total;
        return o;
    };
    auto convFcr  = [&](const Fcr& f) -> FcrH {
        FcrH o; o.fi = conv(f.fi, DMODEL, DFF); o.ff = conv(f.ff, DFF, DMODEL);
        o.ln = f.ln; return o;
    };
    auto convAttn = [&](const Attn& a) -> AttnH {
        AttnH o; o.WQ = conv(a.WQ, DMODEL, DMODEL); o.WK = conv(a.WK, DMODEL, DMODEL);
        o.WV = conv(a.WV, DMODEL, DMODEL); o.WO = conv(a.WO, DMODEL, DMODEL);
        return o;
    };
    FcrH barH[2] = { convFcr(bar_res[0]), convFcr(bar_res[1]) };
    struct EncH { AttnH attn; FcrH fcr; LNP ln; } encH[4];
    for (int i = 0; i < 4; ++i) {
        encH[i].attn = convAttn(enc[i].attn);
        encH[i].fcr  = convFcr(enc[i].fcr);
        encH[i].ln   = enc[i].ln;
    }
    struct DecH { LinH sWV, sWO; AttnH cross; FcrH fcr; LNP ln1, ln2; } decH[4];
    for (int i = 0; i < 4; ++i) {
        decH[i].sWV  = conv(dec[i].self.WV, DMODEL, DMODEL);
        decH[i].sWO  = conv(dec[i].self.WO, DMODEL, DMODEL);
        decH[i].cross = convAttn(dec[i].cross);
        decH[i].fcr  = convFcr(dec[i].fcr);
        decH[i].ln1 = dec[i].ln1; decH[i].ln2 = dec[i].ln2;
    }
    FcrH fcrH[4];
    for (int i = 0; i < 4; ++i) fcrH[i] = convFcr(fcres[i]);

    const long sQb = (long)SEQ * DMODEL;      // 76800 per-batch stride in q/k/v
    const long sSh = (long)SEQ * SLD;         // 22800 per-head score stride
    const long sSb = (long)NHEAD * sSh;       // per-batch score stride

    // ---- bar: per-batch norm, 5->512 embed, 2 fcres, +pos ------------------
    prep_kernel<<<BATCH, 32, 0, stream>>>(mv, qr, x5);
    embed_kernel<<<(int)((TD + 255) / 256), 256, 0, stream>>>(
        x5, bar_init.w, bar_init.b, xbuf, xh, TD);
    run_fcres(stream, xbuf, xh, hh, abuf, barH[0], TOK);
    run_fcres(stream, xbuf, xh, hh, abuf, barH[1], TOK);
    addpos_kernel<<<(int)((TD + 255) / 256), 256, 0, stream>>>(xbuf, xh, pos, TD);

    // ---- encoder x4 --------------------------------------------------------
    for (int l = 0; l < 4; ++l) {
        const AttnH& A = encH[l].attn;
        gemm(stream, xh, DMODEL, 0, 0, A.WQ.wt, DMODEL, 0, 0, 1, A.WQ.b,
             nullptr, qh, DMODEL, 0, 0, TOK, DMODEL, DMODEL, 1, 1, 1.f, 0);
        gemm(stream, xh, DMODEL, 0, 0, A.WK.wt, DMODEL, 0, 0, 1, A.WK.b,
             nullptr, kh, DMODEL, 0, 0, TOK, DMODEL, DMODEL, 1, 1, 1.f, 0);
        gemm(stream, xh, DMODEL, 0, 0, A.WV.wt, DMODEL, 0, 0, 1, A.WV.b,
             nullptr, vh, DMODEL, 0, 0, TOK, DMODEL, DMODEL, 1, 1, 1.f, 0);
        // scores[bh] = (Q_h @ K_h^T)/8 : B operand = K rows (n-major) f16
        gemm(stream, qh, DMODEL, sQb, DK, kh, DMODEL, sQb, DK, 1, nullptr,
             scoresF, nullptr, SLD, sSb, sSh, SEQ, SEQ, DK,
             BATCH * NHEAD, NHEAD, 0.125f, 0);
        softmax150_kernel<<<(BATCH * NHEAD * SEQ + 7) / 8, 256, 0, stream>>>(
            scoresF, probsH, BATCH * NHEAD * SEQ, SLD, SLD);
        // ctx = P @ V : B operand = V k-major f16; out f16 into qh
        gemm(stream, probsH, SLD, sSb, sSh, vh, DMODEL, sQb, DK, 0, nullptr,
             nullptr, qh, DMODEL, sQb, DK, SEQ, DK, SEQ,
             BATCH * NHEAD, NHEAD, 1.f, 0);
        // output projection (fp32, pre-LN)
        gemm(stream, qh, DMODEL, 0, 0, A.WO.wt, DMODEL, 0, 0, 1, A.WO.b,
             abuf, nullptr, DMODEL, 0, 0, TOK, DMODEL, DMODEL, 1, 1, 1.f, 0);
        run_ln(stream, xbuf, xh, abuf, xbuf, encH[l].ln, TOK);
        run_fcres(stream, xbuf, xh, hh, abuf, encH[l].fcr, TOK);
    }

    // ---- decoder x4 (query length 1; self-attn collapses to WO(WV(out))) --
    fill0_kernel<<<(BATCH * DMODEL + 255) / 256, 256, 0, stream>>>(
        dout, douth, (long)BATCH * DMODEL);
    for (int l = 0; l < 4; ++l) {
        gemm(stream, douth, DMODEL, 0, 0, decH[l].sWV.wt, DMODEL, 0, 0, 1,
             decH[l].sWV.b, nullptr, dtmph, DMODEL, 0, 0,
             BATCH, DMODEL, DMODEL, 1, 1, 1.f, 0);
        gemm(stream, dtmph, DMODEL, 0, 0, decH[l].sWO.wt, DMODEL, 0, 0, 1,
             decH[l].sWO.b, dtm2, nullptr, DMODEL, 0, 0,
             BATCH, DMODEL, DMODEL, 1, 1, 1.f, 0);
        run_ln(stream, dout, douth, dtm2, dout, decH[l].ln1, BATCH);

        const AttnH& C = decH[l].cross;
        gemm(stream, douth, DMODEL, 0, 0, C.WQ.wt, DMODEL, 0, 0, 1, C.WQ.b,
             dtmp, nullptr, DMODEL, 0, 0, BATCH, DMODEL, DMODEL, 1, 1, 1.f, 0);
        gemm(stream, xh, DMODEL, 0, 0, C.WK.wt, DMODEL, 0, 0, 1, C.WK.b,
             nullptr, kh, DMODEL, 0, 0, TOK, DMODEL, DMODEL, 1, 1, 1.f, 0);
        gemm(stream, xh, DMODEL, 0, 0, C.WV.wt, DMODEL, 0, 0, 1, C.WV.b,
             nullptr, vh, DMODEL, 0, 0, TOK, DMODEL, DMODEL, 1, 1, 1.f, 0);
        dec_scores_kernel<<<(BATCH * NHEAD * SEQ + 255) / 256, 256, 0, stream>>>(
            dtmp, kh, dsc);
        softmax150_kernel<<<(BATCH * NHEAD + 7) / 8, 256, 0, stream>>>(
            dsc, dscH, BATCH * NHEAD, SEQ, SEQ);
        dec_ctx_kernel<<<(BATCH * DMODEL + 255) / 256, 256, 0, stream>>>(
            dscH, vh, dtmph);
        gemm(stream, dtmph, DMODEL, 0, 0, C.WO.wt, DMODEL, 0, 0, 1, C.WO.b,
             dtmp, nullptr, DMODEL, 0, 0, BATCH, DMODEL, DMODEL, 1, 1, 1.f, 0);
        run_ln(stream, dout, douth, dtmp, dout, decH[l].ln2, BATCH);
        run_fcres(stream, dout, douth, dhh, dtmp, decH[l].fcr, BATCH);
    }

    // ---- trailing fcres x4 + sigmoid head ----------------------------------
    for (int i = 0; i < 4; ++i) run_fcres(stream, dout, douth, dhh, dtmp, fcrH[i], BATCH);
    head_kernel<<<1, 256, 0, stream>>>(dout, fc_disc.w, fc_disc.b, (float*)d_out);
}
